// LigColumnAttention_43130061586665
// MI455X (gfx1250) — matrix operand order repaired and checked by reading, not compiled, not run on hardware
//
#include <hip/hip_runtime.h>
#include <stdint.h>

typedef _Float16 v16h __attribute__((ext_vector_type(16)));
typedef _Float16 v8h  __attribute__((ext_vector_type(8)));
typedef float    v8f  __attribute__((ext_vector_type(8)));
typedef float    v4f  __attribute__((ext_vector_type(4)));

#define WMMA_F32_F16(a,b,c) \
  __builtin_amdgcn_wmma_f32_16x16x32_f16(false,(a),false,(b),(short)0,(c),false,false)

// Build a 16xf16 fragment from two contiguous 8xf16 chunks (each a single
// 16-byte load: ds_load_b128 / global_load_b128).
static __device__ __forceinline__ v16h load_frag_2x8(const _Float16* p0,
                                                     const _Float16* p1) {
  v8h lo = *(const v8h*)p0;
  v8h hi = *(const v8h*)p1;
  v16h r;
#pragma unroll
  for (int j = 0; j < 8; ++j) { r[j] = lo[j]; r[j + 8] = hi[j]; }
  return r;
}

// ---------------------------------------------------------------------------
// Kernel 1: LayerNorm + QKV projection.
//   x   : (16384 tokens, 256) f32, token t = m*32 + i
//   wqkv: (256, 768) f32, column d -> (c = d/24, h' = d%24), h' = s*8 + h
//   out : q/k/v f16 workspaces laid out [i*8+h][m][c]  (head-major)
// Block: 128 threads (4 waves), 64 tokens x 64 cols; k = 256 (8 WMMA steps).
// ---------------------------------------------------------------------------
__global__ void __launch_bounds__(128)
ln_qkv_kernel(const float* __restrict__ lig, const float* __restrict__ lnscale,
              const float* __restrict__ lnbias, const float* __restrict__ wqkv,
              _Float16* __restrict__ qws, _Float16* __restrict__ kws,
              _Float16* __restrict__ vws) {
  extern __shared__ __align__(16) char smem[];
  _Float16* xs  = (_Float16*)smem;   // [64][256] normalized tokens (f16)
  _Float16* wqT = xs + 64 * 256;     // [64][256] w_qkv tile, transposed

  const int tid  = threadIdx.x;
  const int wave = tid >> 5;
  const int lane = tid & 31;
  const int half = lane >> 4;
  const int ln15 = lane & 15;
  const int t0   = blockIdx.x * 64;
  const int col0 = blockIdx.y * 64;

  // ---- LayerNorm: each wave normalizes 16 tokens (32 lanes x 8 channels) ----
  float s8[8], b8[8];
#pragma unroll
  for (int j = 0; j < 8; ++j) {
    s8[j] = lnscale[lane * 8 + j];
    b8[j] = lnbias[lane * 8 + j];
  }
  for (int tt = 0; tt < 16; ++tt) {
    int r = wave * 16 + tt;
    const float* xr = lig + (size_t)(t0 + r) * 256 + lane * 8;
    float xv[8]; float sum = 0.f, sq = 0.f;
#pragma unroll
    for (int j = 0; j < 8; ++j) { float v = xr[j]; xv[j] = v; sum += v; sq += v * v; }
#pragma unroll
    for (int off = 16; off > 0; off >>= 1) {
      sum += __shfl_xor(sum, off);
      sq  += __shfl_xor(sq, off);
    }
    float mu = sum * (1.f / 256.f);
    float rs = rsqrtf(sq * (1.f / 256.f) - mu * mu + 1e-5f);
    _Float16* xd = xs + r * 256 + lane * 8;
#pragma unroll
    for (int j = 0; j < 8; ++j) xd[j] = (_Float16)((xv[j] - mu) * rs * s8[j] + b8[j]);
  }

  // ---- stage w_qkv tile transposed (coalesced global reads, LDS scatter) ----
  // wqT[col][c] = wqkv[c][col0+col]
  for (int idx = tid; idx < 16384; idx += 128) {
    int c   = idx >> 6;      // 0..255
    int col = idx & 63;      // 0..63, consecutive lanes -> consecutive cols
    wqT[col * 256 + c] = (_Float16)wqkv[(size_t)c * 768 + col0 + col];
  }
  __syncthreads();

  // ---- 64x64 GEMM via WMMA: 4 waves x 4 col-tiles x 8 k-steps ----
  v8f acc[4] = {};
  for (int kt = 0; kt < 8; ++kt) {
    const _Float16* ar = xs + (wave * 16 + ln15) * 256 + kt * 32;
    v16h a = load_frag_2x8(ar + half * 16, ar + half * 16 + 8);
#pragma unroll
    for (int ct = 0; ct < 4; ++ct) {
      const _Float16* br = wqT + (ct * 16 + ln15) * 256 + kt * 32 + half * 16;
      acc[ct] = WMMA_F32_F16(a, load_frag_2x8(br, br + 8), acc[ct]);
    }
  }

  // ---- scatter into head-major q/k/v ----
#pragma unroll
  for (int ct = 0; ct < 4; ++ct) {
#pragma unroll
    for (int v = 0; v < 8; ++v) {
      int row = wave * 16 + v + half * 8;
      int t = t0 + row;
      int m = t >> 5, i = t & 31;
      int d  = col0 + ct * 16 + ln15;
      int c  = d / 24;
      int hp = d - c * 24;
      int s = hp >> 3, h = hp & 7;
      _Float16* dst = (s == 0) ? qws : ((s == 1) ? kws : vws);
      dst[((size_t)(i * 8 + h) * 512 + m) * 32 + c] = (_Float16)acc[ct][v];
    }
  }
}

// ---------------------------------------------------------------------------
// Kernel 2: per-head attention. grid = 256 heads x 16 row-strips,
// block = 64 threads (2 waves), each wave owns a 16-row strip.
//   K staged via async global->LDS (ASYNCcnt), V staged transposed.
//   S = (Q*scale) K^T  -> one WMMA per 16x16 tile (K dim = head_dim = 32)
//   softmax over 512 in f32 (vectorized LDS strip), 1/sum folded into O store
//   O = P V            -> 2 col-tiles x 32 k-steps of WMMA
// ---------------------------------------------------------------------------
__global__ void __launch_bounds__(64)
attn_kernel(const _Float16* __restrict__ qws, const _Float16* __restrict__ kws,
            const _Float16* __restrict__ vws, _Float16* __restrict__ ows) {
  extern __shared__ __align__(16) char smem[];
  _Float16* kls = (_Float16*)smem;            // [512][32]  K (row-major)
  _Float16* vT  = (_Float16*)(smem + 32768);  // [32][512]  V transposed
  float*    sls = (float*)(smem + 65536);     // 2 x [16][512] S strips (f32)
  _Float16* pls = (_Float16*)(smem + 131072); // 2 x [16][512] P strips (f16)
  float*    rsc = (float*)(smem + 163840);    // 2 x 16 row 1/sum

  const int tid  = threadIdx.x;
  const int wave = tid >> 5, lane = tid & 31;
  const int half = lane >> 4, ln15 = lane & 15;
  const int ih = blockIdx.x >> 4;
  const int rb = blockIdx.x & 15;
  const int r0 = rb * 32 + wave * 16;
  const size_t hb = (size_t)ih * 512 * 32;

  // ---- K: async global -> LDS, 16 bytes per lane per issue ----
  {
    const _Float16* ksrc = kws + hb;
    for (int idx = tid; idx < 2048; idx += 64) {
      uint32_t lds_off = (uint32_t)(uintptr_t)(kls + idx * 8);
      uint64_t gaddr   = (uint64_t)(uintptr_t)(ksrc + idx * 8);
      asm volatile("global_load_async_to_lds_b128 %0, %1, off"
                   :: "v"(lds_off), "v"(gaddr) : "memory");
    }
  }
  // ---- V: vectorized global reads, transpose scatter into LDS ----
  for (int idx = tid; idx < 2048; idx += 64) {
    int n  = idx >> 2;          // source row 0..511
    int c0 = (idx & 3) * 8;     // channel chunk
    v8h vv = *(const v8h*)(vws + hb + (size_t)n * 32 + c0);
#pragma unroll
    for (int j = 0; j < 8; ++j) vT[(c0 + j) * 512 + n] = vv[j];
  }
  asm volatile("s_wait_asynccnt 0" ::: "memory");
  __syncthreads();

  // ---- Q fragment for this wave's 16 rows, pre-scaled by 1/sqrt(32) ----
  const _Float16* qr = qws + hb + (size_t)(r0 + ln15) * 32;
  v16h qa = load_frag_2x8(qr + half * 16, qr + half * 16 + 8);
#pragma unroll
  for (int j = 0; j < 16; ++j)
    qa[j] = (_Float16)((float)qa[j] * 0.17677669529663687f);

  // ---- S = Q K^T : 32 tiles, one WMMA each (k = 32) ----
  float* sbase = sls + wave * 16 * 512;
  for (int nt = 0; nt < 32; ++nt) {
    const _Float16* br = kls + (nt * 16 + ln15) * 32 + half * 16;
    v8f s = {};
    s = WMMA_F32_F16(qa, load_frag_2x8(br, br + 8), s);
#pragma unroll
    for (int v = 0; v < 8; ++v)
      sbase[(v + half * 8) * 512 + nt * 16 + ln15] = s[v];
  }
  __syncthreads();

  // ---- softmax over 512 (2 lanes per row, float4-vectorized) ----
  {
    int row = lane >> 1, hf = lane & 1;
    const v4f* s4 = (const v4f*)(sbase + row * 512 + hf * 256);
    _Float16* prow = pls + wave * 16 * 512 + row * 512 + hf * 256;
    float mx = -1e30f;
#pragma unroll 4
    for (int j = 0; j < 64; ++j) {
      v4f x = s4[j];
      mx = fmaxf(mx, fmaxf(fmaxf(x[0], x[1]), fmaxf(x[2], x[3])));
    }
    mx = fmaxf(mx, __shfl_xor(mx, 1));
    float sum = 0.f;
    for (int j = 0; j < 32; ++j) {
      v4f a = s4[2 * j], b = s4[2 * j + 1];
      float e[8];
#pragma unroll
      for (int q = 0; q < 4; ++q) { e[q] = __expf(a[q] - mx); e[q + 4] = __expf(b[q] - mx); }
      v8h p;
#pragma unroll
      for (int q = 0; q < 8; ++q) { sum += e[q]; p[q] = (_Float16)e[q]; }
      *(v8h*)(prow + j * 8) = p;
    }
    sum += __shfl_xor(sum, 1);
    if (!hf) rsc[wave * 16 + row] = 1.0f / sum;
  }
  __syncthreads();

  // ---- O = P V : 2 col-tiles x 32 k-steps ----
  v8f o0 = {}, o1 = {};
  const _Float16* pb = pls + wave * 16 * 512;
  for (int kt = 0; kt < 32; ++kt) {
    const _Float16* ar = pb + ln15 * 512 + kt * 32;
    v16h a = load_frag_2x8(ar + half * 16, ar + half * 16 + 8);
    const _Float16* b0 = vT + ln15 * 512 + kt * 32 + half * 16;
    const _Float16* b1 = vT + (16 + ln15) * 512 + kt * 32 + half * 16;
    o0 = WMMA_F32_F16(a, load_frag_2x8(b0, b0 + 8), o0);
    o1 = WMMA_F32_F16(a, load_frag_2x8(b1, b1 + 8), o1);
  }

  // ---- store O into token-major layout [t][c*8 + h], normalized ----
  const int i_ = ih >> 3, h = ih & 7;
#pragma unroll
  for (int v = 0; v < 8; ++v) {
    int M_ = v + half * 8;
    int m = r0 + M_;
    size_t t = (size_t)m * 32 + i_;
    float sc = rsc[wave * 16 + M_];
    ows[t * 256 + ln15 * 8 + h]        = (_Float16)(o0[v] * sc);
    ows[t * 256 + (16 + ln15) * 8 + h] = (_Float16)(o1[v] * sc);
  }
}

// ---------------------------------------------------------------------------
// Kernel 3: output projection  out = O @ w_final + b_final  (f32 output).
// Same tiling as kernel 1: 64 tokens x 64 cols per block, k = 256.
// ---------------------------------------------------------------------------
__global__ void __launch_bounds__(128)
proj_kernel(const _Float16* __restrict__ ows, const float* __restrict__ wfin,
            const float* __restrict__ bfin, float* __restrict__ out) {
  extern __shared__ __align__(16) char smem[];
  _Float16* ot  = (_Float16*)smem;  // [64][256] O tile
  _Float16* wfT = ot + 64 * 256;    // [64][256] w_final tile, transposed

  const int tid = threadIdx.x, wave = tid >> 5, lane = tid & 31;
  const int half = lane >> 4, ln15 = lane & 15;
  const int t0 = blockIdx.x * 64, col0 = blockIdx.y * 64;

  for (int idx = tid; idx < 2048; idx += 128)
    *(v8h*)(ot + idx * 8) = *(const v8h*)(ows + (size_t)t0 * 256 + idx * 8);
  // wfT[col][d] = wfin[d][col0+col]  (coalesced reads, LDS scatter)
  for (int idx = tid; idx < 16384; idx += 128) {
    int d   = idx >> 6;
    int col = idx & 63;
    wfT[col * 256 + d] = (_Float16)wfin[(size_t)d * 256 + col0 + col];
  }
  __syncthreads();

  v8f acc[4] = {};
  for (int kt = 0; kt < 8; ++kt) {
    const _Float16* ar = ot + (wave * 16 + ln15) * 256 + kt * 32;
    v16h a = load_frag_2x8(ar + half * 16, ar + half * 16 + 8);
#pragma unroll
    for (int ct = 0; ct < 4; ++ct) {
      const _Float16* br = wfT + (ct * 16 + ln15) * 256 + kt * 32 + half * 16;
      acc[ct] = WMMA_F32_F16(a, load_frag_2x8(br, br + 8), acc[ct]);
    }
  }

#pragma unroll
  for (int ct = 0; ct < 4; ++ct) {
    int e = col0 + ct * 16 + ln15;
    float bias = bfin[e];
#pragma unroll
    for (int v = 0; v < 8; ++v) {
      int t = t0 + wave * 16 + v + half * 8;
      out[(size_t)t * 256 + e] = acc[ct][v] + bias;
    }
  }
}

// ---------------------------------------------------------------------------
extern "C" void kernel_launch(void* const* d_in, const int* in_sizes, int n_in,
                              void* d_out, int out_size, void* d_ws,
                              size_t ws_size, hipStream_t stream) {
  (void)in_sizes; (void)n_in; (void)out_size; (void)ws_size;
  const float* lig  = (const float*)d_in[0];
  const float* lns  = (const float*)d_in[1];
  const float* lnb  = (const float*)d_in[2];
  const float* wqkv = (const float*)d_in[3];
  const float* wfin = (const float*)d_in[4];
  const float* bfin = (const float*)d_in[5];
  float* out = (float*)d_out;

  // Workspace: q/k/v head-major f16 (8 MB each) + O token-major f16 (8 MB).
  _Float16* qws = (_Float16*)d_ws;
  _Float16* kws = qws + (size_t)256 * 512 * 32;
  _Float16* vws = kws + (size_t)256 * 512 * 32;
  _Float16* ows = vws + (size_t)256 * 512 * 32;

  ln_qkv_kernel<<<dim3(256, 12), 128, 64 * 256 * 2 * 2, stream>>>(
      lig, lns, lnb, wqkv, qws, kws, vws);
  attn_kernel<<<dim3(4096), 64, 163968, stream>>>(qws, kws, vws, ows);
  proj_kernel<<<dim3(256, 4), 128, 64 * 256 * 2 * 2, stream>>>(
      ows, wfin, bfin, out);
}
